// MaskedAttention_67645734912594
// MI455X (gfx1250) — matrix operand order repaired and checked by reading, not compiled, not run on hardware
//
#include <hip/hip_runtime.h>

// MaskedAttention for MI455X (gfx1250, wave32).
// Heavy work = two big GEMMs run as bf16 WMMA (v_wmma_f32_16x16x32_bf16) with
// f32 accumulation; ~206 GFLOP total vs ~0.37GB of HBM traffic (~16us at
// 23.3 TB/s). GEMM K-tiles are staged into LDS with CDNA5 async-to-LDS DMA
// (global_load_async_to_lds_b128, ASYNCcnt) and double-buffered.

typedef unsigned short u16;
typedef __attribute__((ext_vector_type(16))) __bf16 v16bf;
typedef __attribute__((ext_vector_type(8)))  float  v8f;
typedef __attribute__((ext_vector_type(4)))  float  vf4;   // 16B LDS/global chunk
typedef __attribute__((ext_vector_type(4)))  unsigned short v4u16;

union FragBF { v16bf v; vf4 q[2]; };

__device__ __forceinline__ u16 f2bf(float f) {
  unsigned int u = __float_as_uint(f);
  u += 0x7FFFu + ((u >> 16) & 1u);          // round-to-nearest-even
  return (u16)(u >> 16);
}
__device__ __forceinline__ float bf2f(u16 h) {
  return __uint_as_float(((unsigned int)h) << 16);
}

// Async DMA of one 16B chunk from global to LDS (per lane). Generic LDS
// pointers carry the raw LDS byte offset in their low 32 bits on AMDGPU.
__device__ __forceinline__ void async_copy16(const u16* g, u16* l) {
  unsigned ldsOff = (unsigned)(size_t)l;
  asm volatile("global_load_async_to_lds_b128 %0, %1, off"
               :: "v"(ldsOff), "v"((unsigned long long)g) : "memory");
}
__device__ __forceinline__ void wait_async_le2() {
  asm volatile("s_wait_asynccnt 0x2" ::: "memory");
}
__device__ __forceinline__ void wait_async_0() {
  asm volatile("s_wait_asynccnt 0x0" ::: "memory");
}

// ---------------------------------------------------------------- converters
__global__ void k_cvt_bf16(const float* __restrict__ in, u16* __restrict__ out, int n) {
  int i = blockIdx.x * 256 + threadIdx.x;
  if (i < n) out[i] = f2bf(in[i]);
}

// Pack wq(2048x2048), wk(1024x2048), wv(1024x2048) -> Wqkv(4096x2048) bf16.
// Rows are the GEMM N dimension; k stays contiguous (WMMA B operand friendly).
__global__ void k_pack_wqkv(const float* __restrict__ wq, const float* __restrict__ wk,
                            const float* __restrict__ wv, u16* __restrict__ out) {
  int i = blockIdx.x * 256 + threadIdx.x;      // over 4096*2048
  int row = i >> 11;
  int col = i & 2047;
  float v;
  if (row < 2048)       v = wq[i];
  else if (row < 3072)  v = wk[(row - 2048) * 2048 + col];
  else                  v = wv[(row - 3072) * 2048 + col];
  out[i] = f2bf(v);
}

// ---------------------------------------------------------------- bf16 GEMM
// C[M,N] = A[M,K] * B[N,K]^T.  Block tile 128x128, BK=32 (one WMMA K step).
// 256 threads = 8 waves in a 4(M) x 2(N) grid; each wave owns 32x64 = 2x4
// 16x16 f32 accumulators. K tiles double-buffered in LDS via async DMA.
template <bool OUTF32>
__global__ __launch_bounds__(256)
void k_gemm(const u16* __restrict__ A, const u16* __restrict__ B,
            void* __restrict__ C, int M, int N, int K) {
  __shared__ u16 As[2][128][32];   // 2 x 8 KB
  __shared__ u16 Bs[2][128][32];   // 2 x 8 KB

  const int tid  = threadIdx.x;
  const int lane = tid & 31;
  const int wave = tid >> 5;
  const int wm   = wave >> 1;          // 0..3
  const int wn   = wave & 1;           // 0..1
  const int bm   = blockIdx.y * 128;
  const int bn   = blockIdx.x * 128;
  const int lrow = tid >> 1;           // 0..127
  const int lcol = (tid & 1) * 16;     // 0 or 16 halves
  const int hf   = lane >> 4;          // lane half select
  const int r16  = lane & 15;

  v8f acc[2][4] = {};
  const u16* aRow = A + (size_t)(bm + lrow) * K + lcol;
  const u16* bRow = B + (size_t)(bn + lrow) * K + lcol;
  u16* aDst[2] = { &As[0][lrow][lcol], &As[1][lrow][lcol] };
  u16* bDst[2] = { &Bs[0][lrow][lcol], &Bs[1][lrow][lcol] };

  // prime the pipeline: tile 0 -> buffer 0 (2 async instr per wave)
  async_copy16(aRow, aDst[0]);
  async_copy16(bRow, bDst[0]);

  const int nIter = K >> 5;
  for (int t = 0; t < nIter; ++t) {
    const int buf = t & 1;
    if (t + 1 < nIter) {
      // issue next tile into the other buffer, then wait for the *current*
      // tile's pair (async loads complete in order -> <=2 outstanding is safe)
      async_copy16(aRow + (t + 1) * 32, aDst[buf ^ 1]);
      async_copy16(bRow + (t + 1) * 32, bDst[buf ^ 1]);
      wait_async_le2();
    } else {
      wait_async_0();
    }
    __syncthreads();

    FragBF a[2], b[4];
#pragma unroll
    for (int i = 0; i < 2; ++i) {
      // A 16x32 layout: lane half 0 -> K 0..7 & 16..23, half 1 -> K 8..15 & 24..31
      const u16* p = &As[buf][wm * 32 + i * 16 + r16][hf * 8];
      a[i].q[0] = *(const vf4*)p;
      a[i].q[1] = *(const vf4*)(p + 16);
    }
#pragma unroll
    for (int j = 0; j < 4; ++j) {
      // B 32x16 layout: lane = n, half 0 -> K 0..7 & 16..23, half 1 -> K 8..15 & 24..31
      const u16* p = &Bs[buf][wn * 64 + j * 16 + r16][hf * 8];
      b[j].q[0] = *(const vf4*)p;
      b[j].q[1] = *(const vf4*)(p + 16);
    }
#pragma unroll
    for (int i = 0; i < 2; ++i)
#pragma unroll
      for (int j = 0; j < 4; ++j)
        acc[i][j] = __builtin_amdgcn_wmma_f32_16x16x32_bf16(
            false, a[i].v, false, b[j].v, (short)0, acc[i][j], false, false);
    __syncthreads();   // guard the buffer the next iteration will overwrite
  }

  // C/D layout: VGPR r, lane L -> (m = r + 8*(L>>4), n = L&15)
#pragma unroll
  for (int i = 0; i < 2; ++i)
#pragma unroll
    for (int j = 0; j < 4; ++j)
#pragma unroll
      for (int r = 0; r < 8; ++r) {
        int m = bm + wm * 32 + i * 16 + r + 8 * hf;
        int n = bn + wn * 64 + j * 16 + r16;
        if (OUTF32) ((float*)C)[(size_t)m * N + n] = acc[i][j][r];
        else        ((u16*)C)[(size_t)m * N + n]  = f2bf(acc[i][j][r]);
      }
}

// ---------------------------------------------------------------- RoPE
// QKV row layout per position: [Q 16*128 | K 8*128 | V 8*128]. Rotate Q and K.
__global__ void k_rope(u16* __restrict__ qkv, const float* __restrict__ fc,
                       const float* __restrict__ fs, int total) {
  int i = blockIdx.x * 256 + threadIdx.x;
  if (i >= total) return;
  int j    = i & 63;                   // pair index within head dim
  int slot = (i >> 6) % 24;            // 0..15 Q heads, 16..23 K heads
  int pos  = i / (24 * 64);
  int sidx = pos & 2047;               // sequence index
  size_t base = (size_t)pos * 4096 + (slot < 16 ? slot * 128 : 2048 + (slot - 16) * 128);
  float a = bf2f(qkv[base + 2 * j]);
  float b = bf2f(qkv[base + 2 * j + 1]);
  float c = fc[sidx * 64 + j];
  float s = fs[sidx * 64 + j];
  qkv[base + 2 * j]     = f2bf(a * c - b * s);
  qkv[base + 2 * j + 1] = f2bf(a * s + b * c);
}

// ---------------------------------------------------------------- attention
// One wave per (b,s) position. scores(16x16) = Q(16x128) @ K^T(128x16) via
// 4 WMMAs (KV-head repeat folded into the B lane mapping t -> t>>1), softmax
// across 16-lane groups, then P@V through LDS (tiny fraction of FLOPs).
__global__ __launch_bounds__(256)
void k_attn(const u16* __restrict__ qkv, u16* __restrict__ out) {
  __shared__ float Pl[8][16][16];      // 8 KB
  __shared__ u16   Vl[8][1024];        // 16 KB
  const int tid  = threadIdx.x;
  const int lane = tid & 31;
  const int w    = tid >> 5;
  const int pos  = blockIdx.x * 8 + w;
  const int hf   = lane >> 4;
  const int r16  = lane & 15;
  const u16* base = qkv + (size_t)pos * 4096;

  // stage V (8 kv-heads x 128) into LDS: 32 halves per lane
#pragma unroll
  for (int c = 0; c < 4; ++c)
    *(vf4*)&Vl[w][lane * 32 + c * 8] = *(const vf4*)(base + 3072 + lane * 32 + c * 8);

  v8f sc = {};
#pragma unroll
  for (int kc = 0; kc < 4; ++kc) {
    FragBF a, b;
    const u16* pa = base + r16 * 128 + kc * 32 + hf * 8;             // Q, row = head
    a.q[0] = *(const vf4*)pa;
    a.q[1] = *(const vf4*)(pa + 16);
    const u16* pb = base + 2048 + (r16 >> 1) * 128 + kc * 32 + hf * 8;  // K^T, n = t
    b.q[0] = *(const vf4*)pb;
    b.q[1] = *(const vf4*)(pb + 16);
    sc = __builtin_amdgcn_wmma_f32_16x16x32_bf16(false, a.v, false, b.v,
                                                 (short)0, sc, false, false);
  }

  const float scale = 0.08838834764831845f;   // 1/sqrt(128)
#pragma unroll
  for (int r = 0; r < 8; ++r) {
    float v  = sc[r] * scale;
    float mx = v;
#pragma unroll
    for (int off = 1; off < 16; off <<= 1) mx = fmaxf(mx, __shfl_xor(mx, off, 32));
    float e  = __expf(v - mx);
    float sm = e;
#pragma unroll
    for (int off = 1; off < 16; off <<= 1) sm += __shfl_xor(sm, off, 32);
    Pl[w][r + 8 * hf][r16] = e / sm;
  }
  __syncthreads();

  // out[m, d] = sum_t P[m,t] * V[t>>1, d]; lane owns d = lane*4 .. lane*4+3
  float vr[16][4];
#pragma unroll
  for (int t = 0; t < 16; ++t) {
    int kv = t >> 1;
#pragma unroll
    for (int jj = 0; jj < 4; ++jj)
      vr[t][jj] = bf2f(Vl[w][kv * 128 + lane * 4 + jj]);
  }
#pragma unroll
  for (int m = 0; m < 16; ++m) {
    vf4 p4[4];
#pragma unroll
    for (int c = 0; c < 4; ++c) p4[c] = *(vf4*)&Pl[w][m][c * 4];
    const float* pm = (const float*)p4;
    float o[4] = {0.f, 0.f, 0.f, 0.f};
#pragma unroll
    for (int t = 0; t < 16; ++t)
#pragma unroll
      for (int jj = 0; jj < 4; ++jj) o[jj] += pm[t] * vr[t][jj];
    v4u16 st = { f2bf(o[0]), f2bf(o[1]), f2bf(o[2]), f2bf(o[3]) };
    *(v4u16*)&out[(size_t)pos * 2048 + m * 128 + lane * 4] = st;
  }
}

// ---------------------------------------------------------------- launcher
extern "C" void kernel_launch(void* const* d_in, const int* in_sizes, int n_in,
                              void* d_out, int out_size, void* d_ws, size_t ws_size,
                              hipStream_t stream) {
  const float* x  = (const float*)d_in[0];
  const float* fc = (const float*)d_in[1];
  const float* fs = (const float*)d_in[2];
  const float* wq = (const float*)d_in[3];
  const float* wk = (const float*)d_in[4];
  const float* wv = (const float*)d_in[5];
  const float* wo = (const float*)d_in[6];
  float* out = (float*)d_out;
  (void)in_sizes; (void)n_in; (void)out_size; (void)ws_size;

  const int MS   = 4 * 2048;     // b*s = 8192 rows
  const int D    = 2048;
  const int NQKV = 4096;         // 2048 Q + 1024 K + 1024 V

  char* ws = (char*)d_ws;
  u16* Xb   = (u16*)ws;  ws += (size_t)MS * D * 2;      // 32 MB
  u16* Wqkv = (u16*)ws;  ws += (size_t)NQKV * D * 2;    // 16 MB
  u16* Wob  = (u16*)ws;  ws += (size_t)D * D * 2;       //  8 MB
  u16* QKV  = (u16*)ws;  ws += (size_t)MS * NQKV * 2;   // 64 MB
  u16* Ab   = (u16*)ws;  ws += (size_t)MS * D * 2;      // 32 MB

  k_cvt_bf16 <<< (MS * D) / 256,   256, 0, stream >>> (x, Xb, MS * D);
  k_pack_wqkv<<< (NQKV * D) / 256, 256, 0, stream >>> (wq, wk, wv, Wqkv);
  k_cvt_bf16 <<< (D * D) / 256,    256, 0, stream >>> (wo, Wob, D * D);

  k_gemm<false><<< dim3(NQKV / 128, MS / 128), 256, 0, stream >>> (Xb, Wqkv, QKV, MS, NQKV, D);

  int ropeTot = MS * 24 * 64;    // (16 Q heads + 8 K heads) * 64 pairs
  k_rope<<< (ropeTot + 255) / 256, 256, 0, stream >>> (QKV, fc, fs, ropeTot);

  k_attn<<< MS / 8, 256, 0, stream >>> (QKV, Ab);

  k_gemm<true><<< dim3(D / 128, MS / 128), 256, 0, stream >>> (Ab, Wob, out, MS, D, D);
}
